// MultiHeadSelfAttention_90409061581107
// MI455X (gfx1250) — matrix operand order repaired and checked by reading, not compiled, not run on hardware
//
#include <hip/hip_runtime.h>
#include <hip/hip_bf16.h>

typedef _Float16 v16h __attribute__((ext_vector_type(16)));
typedef _Float16 v8h  __attribute__((ext_vector_type(8)));
typedef float    v8f  __attribute__((ext_vector_type(8)));
typedef unsigned int u32x4 __attribute__((ext_vector_type(4)));
typedef unsigned int u32x8 __attribute__((ext_vector_type(8)));

#define D_MODEL 1024
#define N_HEADS 16
#define HEAD_DIM 64
#define SEQ 2048
#define BATCH 2
#define ROWS (BATCH * SEQ) /* 4096 */

// ---------------------------------------------------------------------------
// WMMA fragment helpers (contiguous k order: k = 16*hi + s, wave32)
// ---------------------------------------------------------------------------

// A-fragment 16x32 f16: lane m = lane&15, k-range [hi*16, +16) contiguous
__device__ __forceinline__ v16h load_frag_a(const _Float16* __restrict__ p, int ld,
                                            int row0, int k0, int lane) {
  const int m = lane & 15, hi = lane >> 4;
  const _Float16* r = p + (size_t)(row0 + m) * ld + k0 + hi * 16;
  v8h lo = *(const v8h*)(r);
  v8h hh = *(const v8h*)(r + 8);
  v16h a;
#pragma unroll
  for (int i = 0; i < 8; ++i) { a[i] = lo[i]; a[i + 8] = hh[i]; }
  return a;
}

// B-fragment 32x16 f16 from an N-major source ([N][K] row = n, contiguous k).
__device__ __forceinline__ v16h load_frag_b(const _Float16* __restrict__ pT, int ld,
                                            int n0, int k0, int lane) {
  const int n = lane & 15, hi = lane >> 4;
  return *(const v16h*)(pT + (size_t)(n0 + n) * ld + k0 + hi * 16);
}

// B-fragment from an LDS tile of shape [128 rows][64 k] f16 (row stride 64 elems).
__device__ __forceinline__ v16h load_frag_b_lds(const _Float16* base, int row0,
                                                int kloc, int lane) {
  const int n = lane & 15, hi = lane >> 4;
  return *(const v16h*)(base + (size_t)(row0 + n) * 64 + kloc + hi * 16);
}

__device__ __forceinline__ v8f wmma_f16(v16h a, v16h b, v8f c) {
  return __builtin_amdgcn_wmma_f32_16x16x32_f16(false, a, false, b, (short)0, c,
                                                false, false);
}

// ---------------------------------------------------------------------------
// Tensor Data Mover: DMA a [128 rows x 64 elems] f16 tile of an [1024 x 1024]
// N-major weight matrix into LDS. D# per CDNA5 ISA 8.3/8.4 (2-D, groups 2/3 = 0).
// Issued by one wave; completion tracked with TENSORcnt.
// ---------------------------------------------------------------------------
__device__ __forceinline__ void tdm_load_b_tile(const void* gsrc, unsigned lds_off) {
  const unsigned long long ga = (unsigned long long)(uintptr_t)gsrc;
  u32x4 g0;
  g0[0] = 1u;                                   // count=1, user mode
  g0[1] = lds_off;                              // lds_addr (bytes)
  g0[2] = (unsigned)ga;                         // global_addr[31:0]
  g0[3] = (unsigned)((ga >> 32) & 0x1FFFFFFu)   // global_addr[56:32]
          | (2u << 30);                         // type = 2 ("image")
  u32x8 g1;
  g1[0] = 1u << 16;                             // data_size = 1 (2 bytes)
  g1[1] = (unsigned)(D_MODEL & 0xFFFF) << 16;   // tensor_dim0 lo16 (1024)
  g1[2] = (unsigned)(D_MODEL >> 16)             // tensor_dim0 hi16
          | ((unsigned)(D_MODEL & 0xFFFF) << 16); // tensor_dim1 lo16 (1024)
  g1[3] = (unsigned)(D_MODEL >> 16)             // tensor_dim1 hi16
          | (64u << 16);                        // tile_dim0 = 64
  g1[4] = 128u;                                 // tile_dim1 = 128, tile_dim2 = 0
  g1[5] = (unsigned)D_MODEL;                    // tensor_dim0_stride lo32 (1024)
  g1[6] = 0u;                                   // stride0 hi16 | stride1 lo16
  g1[7] = 0u;
  u32x4 gz = {0u, 0u, 0u, 0u};                  // groups 2/3 unused (2-D tile)
  asm volatile("tensor_load_to_lds %0, %1, %2, %3"
               :: "s"(g0), "s"(g1), "s"(gz), "s"(gz)
               : "memory");
}

// ---------------------------------------------------------------------------
// FP32 -> FP16 conversion / weight transpose
// ---------------------------------------------------------------------------

__global__ __launch_bounds__(256) void cvt_kernel(const float* __restrict__ src,
                                                  _Float16* __restrict__ dst, int n) {
  int i = blockIdx.x * 256 + threadIdx.x;
  if (i < n) dst[i] = (_Float16)src[i];
}

// w: [K][N] row-major f32 (K = N = 1024)  ->  wt: [N][K] f16
__global__ __launch_bounds__(256) void transpose_cvt_kernel(const float* __restrict__ w,
                                                            _Float16* __restrict__ wt) {
  int i = blockIdx.x * 256 + threadIdx.x;
  int k = i >> 10, n = i & (D_MODEL - 1);
  wt[(size_t)n * D_MODEL + k] = (_Float16)w[i];
}

// ---------------------------------------------------------------------------
// Fused QKV projection GEMM: one launch, blockIdx.z selects {Q,K,V}.
//   z==0: Q -> [b,h,n,64], pre-scaled by 1/sqrt(64)
//   z==1: K -> [b,h,n,64]
//   z==2: V -> [b,h,64,n] (transposed so P·V A-fragments load contiguously)
// Block = 8 waves (4M x 2N); wave tile 64x64; block tile 256(M) x 128(N).
// Weight tile [128n x 64k] is TDM-staged into a double-buffered LDS tile;
// 32 WMMAs execute per stage while the next DMA is in flight.
// ---------------------------------------------------------------------------

__global__ __launch_bounds__(256) void gemm_qkv_kernel(
    const _Float16* __restrict__ xh,
    const _Float16* __restrict__ wtq, const _Float16* __restrict__ wtk,
    const _Float16* __restrict__ wtv,
    const float* __restrict__ bq, const float* __restrict__ bk,
    const float* __restrict__ bv,
    _Float16* __restrict__ qh, _Float16* __restrict__ kh,
    _Float16* __restrict__ vth) {
  __shared__ _Float16 bsm[2][128][64];  // 2 x 16 KiB double buffer
  const int lane = threadIdx.x & 31;
  const int wave = threadIdx.x >> 5;
  const int mode = blockIdx.z;
  const _Float16* wt = (mode == 0) ? wtq : (mode == 1) ? wtk : wtv;
  const float* bias = (mode == 0) ? bq : (mode == 1) ? bk : bv;
  _Float16* dst = (mode == 0) ? qh : (mode == 1) ? kh : vth;

  const int m0 = blockIdx.x * 256 + (wave >> 1) * 64;
  const int nw = (wave & 1) * 64;  // wave's N offset inside the block tile

  const unsigned lds0 = (unsigned)(uintptr_t)&bsm[0][0][0];
  const unsigned lds1 = (unsigned)(uintptr_t)&bsm[1][0][0];
  const char* wtb = (const char*)(wt + (size_t)blockIdx.y * 128 * D_MODEL);

  if (wave == 0) tdm_load_b_tile(wtb, lds0);

  v8f c[4][4] = {};
  for (int kc = 0; kc < D_MODEL; kc += 64) {
    const int cur = (kc >> 6) & 1;
    const bool more = (kc + 64) < D_MODEL;
    if (wave == 0 && more)
      tdm_load_b_tile(wtb + (size_t)(kc + 64) * sizeof(_Float16), cur ? lds0 : lds1);
    if (more) __builtin_amdgcn_s_wait_tensorcnt(1);
    else      __builtin_amdgcn_s_wait_tensorcnt(0);
    __syncthreads();
    const _Float16* bbase = &bsm[cur][0][0];
#pragma unroll
    for (int kk = 0; kk < 64; kk += 32) {
      const int k0 = kc + kk;
      __builtin_prefetch(xh + (size_t)(m0 + (lane & 15)) * D_MODEL +
                             ((k0 + 256) & (D_MODEL - 1)), 0, 1);
      v16h a[4], b[4];
#pragma unroll
      for (int i = 0; i < 4; ++i)
        a[i] = load_frag_a(xh, D_MODEL, m0 + i * 16, k0, lane);
#pragma unroll
      for (int i = 0; i < 4; ++i)
        b[i] = load_frag_b_lds(bbase, nw + i * 16, kk, lane);
#pragma unroll
      for (int mi = 0; mi < 4; ++mi)
#pragma unroll
        for (int ni = 0; ni < 4; ++ni)
          c[mi][ni] = wmma_f16(a[mi], b[ni], c[mi][ni]);
    }
    __syncthreads();
  }

  const int hi = lane >> 4;
  const int n0 = blockIdx.y * 128 + nw;
  const float scl = (mode == 0) ? 0.125f : 1.0f;  // 1/sqrt(HEAD_DIM)
#pragma unroll
  for (int nt = 0; nt < 4; ++nt) {
    const int col = n0 + nt * 16 + (lane & 15);
    const float bb = bias[col];
    const int h = col >> 6, d = col & 63;
#pragma unroll
    for (int mt = 0; mt < 4; ++mt) {
#pragma unroll
      for (int j = 0; j < 8; ++j) {
        const int row = m0 + mt * 16 + hi * 8 + j;  // C/D: m = j + 8*hi
        const int b2 = row >> 11, tok = row & (SEQ - 1);
        const float val = (c[mt][nt][j] + bb) * scl;
        size_t addr;
        if (mode == 2)
          addr = ((size_t)((b2 * N_HEADS + h) * HEAD_DIM + d)) * SEQ + tok;
        else
          addr = ((size_t)((b2 * N_HEADS + h) * SEQ + tok)) * HEAD_DIM + d;
        dst[addr] = (_Float16)val;
      }
    }
  }
}

// ---------------------------------------------------------------------------
// Flash attention: one wave per 16-query tile of one (b,h).
// Computes S^T = K·Q^T so the softmax key-reduction is in-lane + one shfl_xor(16),
// then O^T(64x16) += V^T(16x32keys) · P(32keys x 16q) via 4 WMMAs per 32-key step.
// ---------------------------------------------------------------------------

__global__ __launch_bounds__(256) void attn_kernel(const _Float16* __restrict__ qh,
                                                   const _Float16* __restrict__ kh,
                                                   const _Float16* __restrict__ vth,
                                                   _Float16* __restrict__ oh) {
  const int lane = threadIdx.x & 31;
  const int hi = lane >> 4;
  const int gw = blockIdx.x * 8 + (threadIdx.x >> 5);
  const int qt = gw & (SEQ / 16 - 1);  // query tile 0..127
  const int bh = gw >> 7;              // 0..31
  const int b = bh >> 4, h = bh & (N_HEADS - 1);

  const _Float16* Q = qh + (size_t)bh * SEQ * HEAD_DIM;
  const _Float16* K = kh + (size_t)bh * SEQ * HEAD_DIM;
  const _Float16* V = vth + (size_t)bh * HEAD_DIM * SEQ;  // [64][2048]

  // Q tile as B-fragments (n = query, k = head-dim), resident for whole loop
  v16h qf0 = load_frag_b(Q, HEAD_DIM, qt * 16, 0, lane);
  v16h qf1 = load_frag_b(Q, HEAD_DIM, qt * 16, 32, lane);

  v8f o[4] = {};                  // O^T: 4 d-tiles of 16x16
  float m_i = -1e30f, l_i = 0.0f; // per-query running max / sum (dup. on lane pair)

  for (int kt = 0; kt < SEQ / 16; kt += 2) {
    const int kbase = kt * 16;  // 32 keys this iteration
    v8f s0 = {}, s1 = {};
    {
      v16h ka = load_frag_a(K, HEAD_DIM, kbase, 0, lane);
      s0 = wmma_f16(ka, qf0, s0);
      ka = load_frag_a(K, HEAD_DIM, kbase, 32, lane);
      s0 = wmma_f16(ka, qf1, s0);
      ka = load_frag_a(K, HEAD_DIM, kbase + 16, 0, lane);
      s1 = wmma_f16(ka, qf0, s1);
      ka = load_frag_a(K, HEAD_DIM, kbase + 16, 32, lane);
      s1 = wmma_f16(ka, qf1, s1);
    }
    // --- online softmax over keys (rows of S^T) ---
    float tmax = -1e30f;
#pragma unroll
    for (int j = 0; j < 8; ++j) tmax = fmaxf(tmax, fmaxf(s0[j], s1[j]));
    tmax = fmaxf(tmax, __shfl_xor(tmax, 16));
    const float m_new = fmaxf(m_i, tmax);
    const float rescale = __expf(m_i - m_new);
    float e0[8], e1[8], psum = 0.0f;
#pragma unroll
    for (int j = 0; j < 8; ++j) {
      e0[j] = __expf(s0[j] - m_new);  // key = kbase + j + 8*hi
      e1[j] = __expf(s1[j] - m_new);  // key = kbase + 16 + j + 8*hi
      psum += e0[j] + e1[j];
    }
    psum += __shfl_xor(psum, 16);
    l_i = l_i * rescale + psum;
    m_i = m_new;
#pragma unroll
    for (int t = 0; t < 4; ++t)
#pragma unroll
      for (int j = 0; j < 8; ++j) o[t][j] *= rescale;

    // --- build P B-fragment (32 keys x 16 queries): lane k-range = hi*16..+16 ---
    v16h pf;
#pragma unroll
    for (int j = 0; j < 8; ++j) {
      const float send = hi ? e0[j] : e1[j];
      const float recv = __shfl_xor(send, 16);
      pf[j]     = (_Float16)(hi ? recv : e0[j]);  // keys hi*16 + j
      pf[8 + j] = (_Float16)(hi ? e1[j] : recv);  // keys hi*16 + 8 + j
    }
    // --- O^T += V^T · P ---
#pragma unroll
    for (int t = 0; t < 4; ++t) {
      v16h va = load_frag_a(V, SEQ, t * 16, kbase, lane);
      o[t] = wmma_f16(va, pf, o[t]);
    }
  }

  const float inv = 1.0f / l_i;
  const int tok = qt * 16 + (lane & 15);
  _Float16* dst = oh + ((size_t)(b * SEQ + tok)) * D_MODEL + h * HEAD_DIM;
#pragma unroll
  for (int t = 0; t < 4; ++t) {
    v8h pk;  // d = t*16 + hi*8 + j  -> contiguous 8 halves
#pragma unroll
    for (int j = 0; j < 8; ++j) pk[j] = (_Float16)(o[t][j] * inv);
    *(v8h*)(dst + t * 16 + hi * 8) = pk;
  }
}

// ---------------------------------------------------------------------------
// Output projection: out = oh @ wo + bo, f32 result. Same TDM-staged tiling.
// ---------------------------------------------------------------------------

__global__ __launch_bounds__(256) void gemm_out_kernel(const _Float16* __restrict__ ah,
                                                       const _Float16* __restrict__ wt,
                                                       const float* __restrict__ bias,
                                                       float* __restrict__ out) {
  __shared__ _Float16 bsm[2][128][64];
  const int lane = threadIdx.x & 31;
  const int wave = threadIdx.x >> 5;
  const int m0 = blockIdx.x * 256 + (wave >> 1) * 64;
  const int nw = (wave & 1) * 64;

  const unsigned lds0 = (unsigned)(uintptr_t)&bsm[0][0][0];
  const unsigned lds1 = (unsigned)(uintptr_t)&bsm[1][0][0];
  const char* wtb = (const char*)(wt + (size_t)blockIdx.y * 128 * D_MODEL);

  if (wave == 0) tdm_load_b_tile(wtb, lds0);

  v8f c[4][4] = {};
  for (int kc = 0; kc < D_MODEL; kc += 64) {
    const int cur = (kc >> 6) & 1;
    const bool more = (kc + 64) < D_MODEL;
    if (wave == 0 && more)
      tdm_load_b_tile(wtb + (size_t)(kc + 64) * sizeof(_Float16), cur ? lds0 : lds1);
    if (more) __builtin_amdgcn_s_wait_tensorcnt(1);
    else      __builtin_amdgcn_s_wait_tensorcnt(0);
    __syncthreads();
    const _Float16* bbase = &bsm[cur][0][0];
#pragma unroll
    for (int kk = 0; kk < 64; kk += 32) {
      const int k0 = kc + kk;
      __builtin_prefetch(ah + (size_t)(m0 + (lane & 15)) * D_MODEL +
                             ((k0 + 256) & (D_MODEL - 1)), 0, 1);
      v16h a[4], b[4];
#pragma unroll
      for (int i = 0; i < 4; ++i)
        a[i] = load_frag_a(ah, D_MODEL, m0 + i * 16, k0, lane);
#pragma unroll
      for (int i = 0; i < 4; ++i)
        b[i] = load_frag_b_lds(bbase, nw + i * 16, kk, lane);
#pragma unroll
      for (int mi = 0; mi < 4; ++mi)
#pragma unroll
        for (int ni = 0; ni < 4; ++ni)
          c[mi][ni] = wmma_f16(a[mi], b[ni], c[mi][ni]);
    }
    __syncthreads();
  }

  const int hi = lane >> 4;
  const int n0 = blockIdx.y * 128 + nw;
#pragma unroll
  for (int nt = 0; nt < 4; ++nt) {
    const int col = n0 + nt * 16 + (lane & 15);
    const float bb = bias[col];
#pragma unroll
    for (int mt = 0; mt < 4; ++mt) {
#pragma unroll
      for (int j = 0; j < 8; ++j) {
        const int row = m0 + mt * 16 + hi * 8 + j;
        out[(size_t)row * D_MODEL + col] = c[mt][nt][j] + bb;
      }
    }
  }
}

// ---------------------------------------------------------------------------

extern "C" void kernel_launch(void* const* d_in, const int* in_sizes, int n_in,
                              void* d_out, int out_size, void* d_ws, size_t ws_size,
                              hipStream_t stream) {
  (void)in_sizes; (void)n_in; (void)out_size; (void)ws_size;
  const float* x  = (const float*)d_in[0];
  const float* wq = (const float*)d_in[1];
  const float* bq = (const float*)d_in[2];
  const float* wk = (const float*)d_in[3];
  const float* bk = (const float*)d_in[4];
  const float* wv = (const float*)d_in[5];
  const float* bv = (const float*)d_in[6];
  const float* wo = (const float*)d_in[7];
  const float* bo = (const float*)d_in[8];
  float* out = (float*)d_out;

  char* ws = (char*)d_ws;
  _Float16* xh  = (_Float16*)(ws);                        // 8 MiB  [4096][1024]
  _Float16* wtq = (_Float16*)(ws + ((size_t)8  << 20));   // 2 MiB  [N][K]
  _Float16* wtk = (_Float16*)(ws + ((size_t)10 << 20));
  _Float16* wtv = (_Float16*)(ws + ((size_t)12 << 20));
  _Float16* wto = (_Float16*)(ws + ((size_t)14 << 20));
  _Float16* qh  = (_Float16*)(ws + ((size_t)16 << 20));   // 8 MiB  [b,h,n,64]
  _Float16* kh  = (_Float16*)(ws + ((size_t)24 << 20));   // 8 MiB  [b,h,n,64]
  _Float16* vth = (_Float16*)(ws + ((size_t)32 << 20));   // 8 MiB  [b,h,64,n]
  _Float16* oh  = (_Float16*)(ws + ((size_t)40 << 20));   // 8 MiB  [4096][1024]

  cvt_kernel<<<(ROWS * D_MODEL) / 256, 256, 0, stream>>>(x, xh, ROWS * D_MODEL);
  transpose_cvt_kernel<<<(D_MODEL * D_MODEL) / 256, 256, 0, stream>>>(wq, wtq);
  transpose_cvt_kernel<<<(D_MODEL * D_MODEL) / 256, 256, 0, stream>>>(wk, wtk);
  transpose_cvt_kernel<<<(D_MODEL * D_MODEL) / 256, 256, 0, stream>>>(wv, wtv);
  transpose_cvt_kernel<<<(D_MODEL * D_MODEL) / 256, 256, 0, stream>>>(wo, wto);

  dim3 gqkv(ROWS / 256, D_MODEL / 128, 3);
  gemm_qkv_kernel<<<gqkv, 256, 0, stream>>>(xh, wtq, wtk, wtv, bq, bk, bv,
                                            qh, kh, vth);

  attn_kernel<<<(BATCH * N_HEADS * (SEQ / 16)) / 8, 256, 0, stream>>>(qh, kh, vth, oh);

  dim3 gout(ROWS / 256, D_MODEL / 128);
  gemm_out_kernel<<<gout, 256, 0, stream>>>(oh, wto, bo, out);
}
